// Informer_56444460204576
// MI455X (gfx1250) — matrix operand order repaired and checked by reading, compiled, not run on hardware
//
#include <hip/hip_runtime.h>
#include <hip/hip_bf16.h>

typedef _Float16 v16h __attribute__((ext_vector_type(16)));
typedef _Float16 v8h  __attribute__((ext_vector_type(8)));
typedef float    v8f  __attribute__((ext_vector_type(8)));

#define DMODEL 512
#define NHEAD  8
#define DHEAD  64
#define NB     8   // batch

static __device__ __forceinline__ v8f wmma32(v16h a, v16h b, v8f c) {
  return __builtin_amdgcn_wmma_f32_16x16x32_f16(false, a, false, b, (short)0, c, false, false);
}
static __device__ __forceinline__ v16h pack16(v8h lo, v8h hi) {
  v16h r;
#pragma unroll
  for (int i = 0; i < 8; i++) { r[i] = lo[i]; r[i + 8] = hi[i]; }
  return r;
}

// ---------------------------------------------------------------------------
// Tiled WMMA GEMM:  C[M,N] = act( A[M,K] * W[K,N] + bias )  (optional +=C)
// BM=128 BN=128 BK=32, 8 waves (4x2), each wave 32x64 via 8 wmma tiles.
// Double-buffered LDS with register staging; one barrier per K-step.
// Requires M%128==0, N%128==0, K%32==0 (true for all call sites).
// ---------------------------------------------------------------------------
#define GBM 128
#define GBN 128
#define GBK 32
#define ALD 40   // halves per A-lds row (32 + 8 pad)
#define BLD 40   // halves per Bt-lds row

template <int ACT, bool ACCUM, bool HASBIAS>
__global__ __launch_bounds__(256) void inf_gemm(
    const float* __restrict__ A, const float* __restrict__ W,
    const float* __restrict__ bias, float* __restrict__ C,
    int M, int N, int K)
{
  __shared__ __align__(16) _Float16 Af[2][GBM * ALD];
  __shared__ __align__(16) _Float16 Bt[2][GBN * BLD];

  const int tid   = threadIdx.x;
  const int lane  = tid & 31;
  const int wave  = tid >> 5;
  const int waveM = wave >> 1;   // 0..3
  const int waveN = wave & 1;    // 0..1
  const int ln16  = lane & 15;
  const int hsel  = (lane < 16) ? 0 : 1;
  const int tileM = blockIdx.y * GBM;
  const int tileN = blockIdx.x * GBN;

  const int rA = tid >> 1;          // 0..127
  const int cA = (tid & 1) * 16;    // 0,16
  const int kB = tid >> 3;          // 0..31
  const int nB = (tid & 7) * 16;    // 0..112

  float aR[16], bR[16];
  v8f acc[2][4];
#pragma unroll
  for (int mt = 0; mt < 2; mt++)
#pragma unroll
    for (int nt = 0; nt < 4; nt++)
#pragma unroll
      for (int v = 0; v < 8; v++) acc[mt][nt][v] = 0.0f;

  const int nIter = K / GBK;

  // ---- prologue: load + stage tile 0 into buffer 0 ----
  {
    const float* as = A + (size_t)(tileM + rA) * K + cA;
#pragma unroll
    for (int i = 0; i < 16; i++) aR[i] = as[i];
    const float* bs = W + (size_t)kB * N + tileN + nB;
#pragma unroll
    for (int i = 0; i < 16; i++) bR[i] = bs[i];

    v8h h0, h1;
#pragma unroll
    for (int i = 0; i < 8; i++) { h0[i] = (_Float16)aR[i]; h1[i] = (_Float16)aR[8 + i]; }
    *(v8h*)&Af[0][rA * ALD + cA]     = h0;
    *(v8h*)&Af[0][rA * ALD + cA + 8] = h1;
#pragma unroll
    for (int i = 0; i < 16; i++) Bt[0][(nB + i) * BLD + kB] = (_Float16)bR[i];
  }
  __syncthreads();

  for (int it = 0; it < nIter; ++it) {
    const int cur = it & 1;
    const bool hasNext = (it + 1) < nIter;

    // issue next tile's global loads early (latency hidden behind wmma)
    if (hasNext) {
      int k0 = (it + 1) * GBK;
      const float* as = A + (size_t)(tileM + rA) * K + k0 + cA;
#pragma unroll
      for (int i = 0; i < 16; i++) aR[i] = as[i];
      const float* bs = W + (size_t)(k0 + kB) * N + tileN + nB;
#pragma unroll
      for (int i = 0; i < 16; i++) bR[i] = bs[i];
    }

    // compute from current buffer
    v16h afrag[2], bfrag[4];
#pragma unroll
    for (int mt = 0; mt < 2; mt++) {
      int m = waveM * 32 + mt * 16 + ln16;
      v8h lo = *(const v8h*)&Af[cur][m * ALD + hsel * 8];
      v8h hi = *(const v8h*)&Af[cur][m * ALD + 16 + hsel * 8];
      afrag[mt] = pack16(lo, hi);
    }
#pragma unroll
    for (int nt = 0; nt < 4; nt++) {
      int n = waveN * 64 + nt * 16 + ln16;
      const v8h* p = (const v8h*)&Bt[cur][n * BLD + hsel * 8];
      bfrag[nt] = pack16(p[0], p[2]);
    }
#pragma unroll
    for (int mt = 0; mt < 2; mt++)
#pragma unroll
      for (int nt = 0; nt < 4; nt++)
        acc[mt][nt] = wmma32(afrag[mt], bfrag[nt], acc[mt][nt]);

    // stage next tile into alternate buffer
    if (hasNext) {
      const int nxt = 1 - cur;
      v8h h0, h1;
#pragma unroll
      for (int i = 0; i < 8; i++) { h0[i] = (_Float16)aR[i]; h1[i] = (_Float16)aR[8 + i]; }
      *(v8h*)&Af[nxt][rA * ALD + cA]     = h0;
      *(v8h*)&Af[nxt][rA * ALD + cA + 8] = h1;
#pragma unroll
      for (int i = 0; i < 16; i++) Bt[nxt][(nB + i) * BLD + kB] = (_Float16)bR[i];
    }
    __syncthreads();
  }

  // ---- epilogue (fully branch-free per template instantiation) ----
#pragma unroll
  for (int mt = 0; mt < 2; mt++)
#pragma unroll
    for (int nt = 0; nt < 4; nt++) {
      int n = tileN + waveN * 64 + nt * 16 + ln16;
      float bv = HASBIAS ? bias[n] : 0.0f;
#pragma unroll
      for (int v = 0; v < 8; v++) {
        int m = tileM + waveM * 32 + mt * 16 + v + hsel * 8;
        size_t off = (size_t)m * N + n;
        float val = acc[mt][nt][v] + bv;
        if (ACCUM) val += C[off];
        if (ACT == 1) val = 0.5f * val * (1.0f + erff(val * 0.70710678f));
        C[off] = val;
      }
    }
}

// ---------------------------------------------------------------------------
// Flash attention with WMMA. Q[B,Lq,512],K/V[B,Lk,512] merged-head layout,
// head selected by blockIdx.y. 128 Q rows / block (8 waves x 16 rows),
// KV streamed in 32-wide tiles. Online softmax, scale=1/8.
// ---------------------------------------------------------------------------
template <int CAUSAL>
__global__ __launch_bounds__(256) void inf_flash(
    const float* __restrict__ Q, const float* __restrict__ Kp,
    const float* __restrict__ Vp, float* __restrict__ O,
    int Lq, int Lk)
{
  __shared__ __align__(16) _Float16 Qs[128 * 72];
  __shared__ __align__(16) _Float16 Ks[32 * 72];
  __shared__ __align__(16) _Float16 Vt[64 * 40];
  __shared__ __align__(16) _Float16 Ps[8 * 16 * 40];

  const int tid  = threadIdx.x;
  const int lane = tid & 31;
  const int wave = tid >> 5;
  const int ln16 = lane & 15;
  const int hsel = (lane < 16) ? 0 : 1;
  const int b = blockIdx.z;
  const int h = blockIdx.y;
  const int qBase = blockIdx.x * 128;

  // Load Q tile (128x64) fp32 -> f16, 16B LDS stores
  {
    int r = tid >> 1;
    int c = (tid & 1) * 32;
    const float* src = Q + (size_t)(b * Lq + qBase + r) * DMODEL + h * DHEAD + c;
    _Float16* dst = &Qs[r * 72 + c];
#pragma unroll
    for (int j = 0; j < 4; j++) {
      v8h hv;
#pragma unroll
      for (int i = 0; i < 8; i++) hv[i] = (_Float16)src[j * 8 + i];
      *(v8h*)&dst[j * 8] = hv;
    }
  }
  __syncthreads();

  v8f oacc[4];
#pragma unroll
  for (int t = 0; t < 4; t++)
#pragma unroll
    for (int v = 0; v < 8; v++) oacc[t][v] = 0.0f;
  float mrow[8], lrow[8];
#pragma unroll
  for (int v = 0; v < 8; v++) { mrow[v] = -3.0e38f; lrow[v] = 0.0f; }

  const int sEnd = CAUSAL ? ((qBase + 128 < Lk) ? qBase + 128 : Lk) : Lk;
  const int sRow = tid >> 3;
  const int sCol = (tid & 7) * 8;

  for (int s0 = 0; s0 < sEnd; s0 += 32) {
    // K tile rows -> Ks[s][d] (16B stores), V tile -> Vt[d][s] (transposed)
    {
      const float* ks = Kp + (size_t)(b * Lk + s0 + sRow) * DMODEL + h * DHEAD + sCol;
      const float* vs = Vp + (size_t)(b * Lk + s0 + sRow) * DMODEL + h * DHEAD + sCol;
      if (s0 + 32 < sEnd) {  // prefetch next KV tile (global_prefetch_b8)
        __builtin_prefetch(ks + 32 * DMODEL, 0, 1);
        __builtin_prefetch(vs + 32 * DMODEL, 0, 1);
      }
      v8h hv;
#pragma unroll
      for (int i = 0; i < 8; i++) hv[i] = (_Float16)ks[i];
      *(v8h*)&Ks[sRow * 72 + sCol] = hv;
#pragma unroll
      for (int i = 0; i < 8; i++) Vt[(sCol + i) * 40 + sRow] = (_Float16)vs[i];
    }
    __syncthreads();

    // S = Q * K^T   (16x32 tile per wave, 2 C-tiles of 16x16)
    v8f S[2];
#pragma unroll
    for (int nt = 0; nt < 2; nt++)
#pragma unroll
      for (int v = 0; v < 8; v++) S[nt][v] = 0.0f;

#pragma unroll
    for (int kk = 0; kk < 2; kk++) {
      int m = wave * 16 + ln16;
      v8h alo = *(const v8h*)&Qs[m * 72 + kk * 32 + hsel * 8];
      v8h ahi = *(const v8h*)&Qs[m * 72 + kk * 32 + 16 + hsel * 8];
      v16h af = pack16(alo, ahi);
#pragma unroll
      for (int nt = 0; nt < 2; nt++) {
        int s = nt * 16 + ln16;
        const v8h* bp = (const v8h*)&Ks[s * 72 + kk * 32 + hsel * 8];
        v16h bf = pack16(bp[0], bp[2]);
        S[nt] = wmma32(af, bf, S[nt]);
      }
    }

    // Online softmax update (rows v and v+8 split across lane halves)
    float corr[8];
#pragma unroll
    for (int v = 0; v < 8; v++) {
      int mg = qBase + wave * 16 + v + hsel * 8;
      float s0v = S[0][v] * 0.125f;
      float s1v = S[1][v] * 0.125f;
      if (CAUSAL) {
        if (s0 + ln16 > mg)      s0v = -3.0e38f;
        if (s0 + 16 + ln16 > mg) s1v = -3.0e38f;
      }
      float t = fmaxf(s0v, s1v);
      t = fmaxf(t, __shfl_xor(t, 1));
      t = fmaxf(t, __shfl_xor(t, 2));
      t = fmaxf(t, __shfl_xor(t, 4));
      t = fmaxf(t, __shfl_xor(t, 8));
      float mn = fmaxf(mrow[v], t);
      float c  = __expf(mrow[v] - mn);
      mrow[v] = mn;
      corr[v] = c;
      float p0 = (s0v <= -1.0e38f) ? 0.0f : __expf(s0v - mn);
      float p1 = (s1v <= -1.0e38f) ? 0.0f : __expf(s1v - mn);
      S[0][v] = p0; S[1][v] = p1;
      float rs = p0 + p1;
      rs += __shfl_xor(rs, 1);
      rs += __shfl_xor(rs, 2);
      rs += __shfl_xor(rs, 4);
      rs += __shfl_xor(rs, 8);
      lrow[v] = lrow[v] * c + rs;
    }
#pragma unroll
    for (int t = 0; t < 4; t++)
#pragma unroll
      for (int v = 0; v < 8; v++) oacc[t][v] *= corr[v];

    // Stage P (C layout -> A layout) through per-wave LDS
    _Float16* pw = &Ps[wave * 16 * 40];
#pragma unroll
    for (int nt = 0; nt < 2; nt++)
#pragma unroll
      for (int v = 0; v < 8; v++)
        pw[(v + hsel * 8) * 40 + nt * 16 + ln16] = (_Float16)S[nt][v];

    v8h plo = *(const v8h*)&pw[ln16 * 40 + hsel * 8];
    v8h phi = *(const v8h*)&pw[ln16 * 40 + 16 + hsel * 8];
    v16h pf = pack16(plo, phi);

#pragma unroll
    for (int t = 0; t < 4; t++) {
      int d = t * 16 + ln16;
      const v8h* vp = (const v8h*)&Vt[d * 40 + hsel * 8];
      v16h vf = pack16(vp[0], vp[2]);
      oacc[t] = wmma32(pf, vf, oacc[t]);
    }
    __syncthreads();
  }

  float inv[8];
#pragma unroll
  for (int v = 0; v < 8; v++) inv[v] = 1.0f / lrow[v];
#pragma unroll
  for (int t = 0; t < 4; t++)
#pragma unroll
    for (int v = 0; v < 8; v++) {
      int m = qBase + wave * 16 + v + hsel * 8;
      O[(size_t)(b * Lq + m) * DMODEL + h * DHEAD + t * 16 + ln16] =
          oacc[t][v] * inv[v];
    }
}

// ---------------------------------------------------------------------------
// Data embedding: circular conv3 (Cin=7) + positional + temporal sinusoids
// ---------------------------------------------------------------------------
__global__ void inf_embed(const float* __restrict__ x, const int* __restrict__ mark,
                          const float* __restrict__ w, float* __restrict__ out,
                          int L, int Cin)
{
  size_t idx = (size_t)blockIdx.x * blockDim.x + threadIdx.x;
  size_t total = (size_t)NB * L * DMODEL;
  if (idx >= total) return;
  int d = idx & (DMODEL - 1);
  int l = (int)((idx >> 9) % L);
  int b = (int)(idx / ((size_t)L * DMODEL));

  float val = 0.0f;
#pragma unroll
  for (int t = 0; t < 3; t++) {
    int ls = (l - 1 + t + L) % L;
    const float* xr = x + (size_t)(b * L + ls) * Cin;
    const float* wr = w + (size_t)t * Cin * DMODEL + d;
    for (int c = 0; c < 7; c++) val += xr[c] * wr[(size_t)c * DMODEL];
  }
  float freq = __expf(-0.0179889460f * (float)(d & ~1));
  float pe = (d & 1) ? __cosf((float)l * freq) : __sinf((float)l * freq);
  float tmp = 0.0f;
  const int* mk = mark + (size_t)(b * L + l) * 4;
#pragma unroll
  for (int j = 0; j < 4; j++) {
    float pos = (float)mk[j];
    tmp += (d & 1) ? __cosf(pos * freq) : __sinf(pos * freq);
  }
  out[idx] = val + pe + tmp;
}

// ---------------------------------------------------------------------------
// out = LayerNorm(x (+ res)) over last dim 512; one block per row
// ---------------------------------------------------------------------------
template <bool HASRES>
__global__ __launch_bounds__(256) void inf_ln(
    const float* __restrict__ x, const float* __restrict__ res,
    const float* __restrict__ g, const float* __restrict__ bt,
    float* __restrict__ out)
{
  __shared__ float rs[8], rq[8];
  int row = blockIdx.x;
  int tid = threadIdx.x;
  int lane = tid & 31, wave = tid >> 5;
  size_t base = (size_t)row * DMODEL;
  float v0 = x[base + tid], v1 = x[base + tid + 256];
  if (HASRES) { v0 += res[base + tid]; v1 += res[base + tid + 256]; }
  float s = v0 + v1, q = v0 * v0 + v1 * v1;
#pragma unroll
  for (int off = 16; off >= 1; off >>= 1) {
    s += __shfl_xor(s, off);
    q += __shfl_xor(q, off);
  }
  if (lane == 0) { rs[wave] = s; rq[wave] = q; }
  __syncthreads();
  if (tid == 0) {
    float a = 0.0f, c = 0.0f;
    for (int i = 0; i < 8; i++) { a += rs[i]; c += rq[i]; }
    rs[0] = a; rq[0] = c;
  }
  __syncthreads();
  float mean = rs[0] * (1.0f / DMODEL);
  float var  = rq[0] * (1.0f / DMODEL) - mean * mean;
  float rstd = rsqrtf(var + 1e-5f);
  out[base + tid]       = (v0 - mean) * rstd * g[tid]       + bt[tid];
  out[base + tid + 256] = (v1 - mean) * rstd * g[tid + 256] + bt[tid + 256];
}

// ---------------------------------------------------------------------------
// BatchNorm stats over rows (per channel), then BN+ELU, then maxpool s2k3
// ---------------------------------------------------------------------------
__global__ __launch_bounds__(256) void inf_bnstats(
    const float* __restrict__ y, float* __restrict__ stats, int rows)
{
  __shared__ float rs[8], rq[8];
  int ch = blockIdx.x;
  int tid = threadIdx.x;
  int lane = tid & 31, wave = tid >> 5;
  float s = 0.0f, q = 0.0f;
  for (int r = tid; r < rows; r += 256) {
    float v = y[(size_t)r * DMODEL + ch];
    s += v; q += v * v;
  }
#pragma unroll
  for (int off = 16; off >= 1; off >>= 1) {
    s += __shfl_xor(s, off);
    q += __shfl_xor(q, off);
  }
  if (lane == 0) { rs[wave] = s; rq[wave] = q; }
  __syncthreads();
  if (tid == 0) {
    float a = 0.0f, c = 0.0f;
    for (int i = 0; i < 8; i++) { a += rs[i]; c += rq[i]; }
    stats[ch] = a; stats[DMODEL + ch] = c;
  }
}

__global__ void inf_bnelu(float* __restrict__ y, const float* __restrict__ stats,
                          const float* __restrict__ g, const float* __restrict__ bt,
                          int rows)
{
  size_t idx = (size_t)blockIdx.x * blockDim.x + threadIdx.x;
  size_t total = (size_t)rows * DMODEL;
  if (idx >= total) return;
  int ch = idx & (DMODEL - 1);
  float inv = 1.0f / (float)rows;
  float mean = stats[ch] * inv;
  float var  = stats[DMODEL + ch] * inv - mean * mean;
  float v = (y[idx] - mean) * rsqrtf(var + 1e-5f) * g[ch] + bt[ch];
  y[idx] = (v > 0.0f) ? v : (__expf(v) - 1.0f);
}

__global__ void inf_maxpool(float* __restrict__ out, const float* __restrict__ z, int L)
{
  int Lo = L >> 1;
  size_t idx = (size_t)blockIdx.x * blockDim.x + threadIdx.x;
  size_t total = (size_t)NB * Lo * DMODEL;
  if (idx >= total) return;
  int d = idx & (DMODEL - 1);
  int lo = (int)((idx >> 9) % Lo);
  int b = (int)(idx / ((size_t)Lo * DMODEL));
  size_t base = (size_t)(b * L + 2 * lo) * DMODEL + d;
  float m = z[base];
  if (2 * lo - 1 >= 0) m = fmaxf(m, z[base - DMODEL]);
  if (2 * lo + 1 < L)  m = fmaxf(m, z[base + DMODEL]);
  out[idx] = m;
}

__global__ void inf_shift(float* __restrict__ dst, const float* __restrict__ src,
                          int L, int sh)
{
  size_t idx = (size_t)blockIdx.x * blockDim.x + threadIdx.x;
  size_t total = (size_t)NB * L * DMODEL;
  if (idx >= total) return;
  int d = idx & (DMODEL - 1);
  int l = (int)((idx >> 9) % L);
  int b = (int)(idx / ((size_t)L * DMODEL));
  int ls = (l + sh + L) % L;
  dst[idx] = src[(size_t)(b * L + ls) * DMODEL + d];
}

__global__ void inf_proj(const float* __restrict__ x, const float* __restrict__ pw,
                         const float* __restrict__ pb, float* __restrict__ out)
{
  int idx = blockIdx.x * blockDim.x + threadIdx.x;
  if (idx >= NB * 256 * 7) return;
  int c = idx % 7;
  int t = (idx / 7) % 256;
  int b = idx / (7 * 256);
  const float* xr = x + (size_t)(b * 512 + 256 + t) * DMODEL;
  float s = pb[c];
  for (int k = 0; k < DMODEL; k++) s += xr[k] * pw[(size_t)k * 7 + c];
  out[idx] = s;
}

// ---------------------------------------------------------------------------
// Host orchestration
// ---------------------------------------------------------------------------
extern "C" void kernel_launch(void* const* d_in, const int* in_sizes, int n_in,
                              void* d_out, int out_size, void* d_ws, size_t ws_size,
                              hipStream_t stream) {
  (void)in_sizes; (void)n_in; (void)out_size; (void)ws_size;

  const float* x_enc    = (const float*)d_in[0];
  const int*   mark_enc = (const int*)  d_in[1];
  const float* x_dec    = (const float*)d_in[2];
  const int*   mark_dec = (const int*)  d_in[3];
  const float* Pp[58];
  for (int i = 4; i < 58; i++) Pp[i] = (const float*)d_in[i];

  const size_t DD = (size_t)DMODEL * DMODEL;   // 512*512
  const size_t M1 = 1024 * 1024;

  float* ws   = (float*)d_ws;
  float* X    = ws;                // 4M floats
  float* Qb   = ws + 4 * M1;
  float* Kb   = ws + 8 * M1;
  float* Vb   = ws + 12 * M1;
  float* T1   = ws + 16 * M1;
  float* ENC  = ws + 20 * M1;      // 1M
  float* XD   = ws + 21 * M1;      // 2M
  float* STAT = ws + 23 * M1;      // 1K

  auto gemm = [&](const float* A, const float* W, const float* bias, float* C,
                  int M, int N, int K, int act, int accum) {
    dim3 g(N / GBN, M / GBM);
    if (accum)
      inf_gemm<0, true, false><<<g, 256, 0, stream>>>(A, W, nullptr, C, M, N, K);
    else if (act == 1)
      inf_gemm<1, false, true><<<g, 256, 0, stream>>>(A, W, bias, C, M, N, K);
    else
      inf_gemm<0, false, true><<<g, 256, 0, stream>>>(A, W, bias, C, M, N, K);
  };
  auto flash = [&](const float* Q, const float* K, const float* V, float* O,
                   int Lq, int Lk, int causal) {
    dim3 g(Lq / 128, NHEAD, NB);
    if (causal) inf_flash<1><<<g, 256, 0, stream>>>(Q, K, V, O, Lq, Lk);
    else        inf_flash<0><<<g, 256, 0, stream>>>(Q, K, V, O, Lq, Lk);
  };
  auto ln = [&](int rows, const float* x, const float* res, const float* g,
                const float* bt, float* out) {
    if (res) inf_ln<true><<<rows, 256, 0, stream>>>(x, res, g, bt, out);
    else     inf_ln<false><<<rows, 256, 0, stream>>>(x, nullptr, g, bt, out);
  };
  auto elts = [&](size_t n) { return (unsigned)((n + 255) / 256); };

  // ============================== Encoder ==============================
  {
    int L = 1024;
    inf_embed<<<elts((size_t)NB * L * DMODEL), 256, 0, stream>>>(
        x_enc, mark_enc, Pp[4], X, L, 7);

    for (int l = 0; l < 3; l++) {
      int M = NB * L;
      gemm(X, Pp[6]  + l * DD, Pp[7]  + l * DMODEL, Qb, M, DMODEL, DMODEL, 0, 0);
      gemm(X, Pp[8]  + l * DD, Pp[9]  + l * DMODEL, Kb, M, DMODEL, DMODEL, 0, 0);
      gemm(X, Pp[10] + l * DD, Pp[11] + l * DMODEL, Vb, M, DMODEL, DMODEL, 0, 0);
      flash(Qb, Kb, Vb, T1, L, L, 0);
      gemm(T1, Pp[12] + l * DD, Pp[13] + l * DMODEL, Qb, M, DMODEL, DMODEL, 0, 0);
      ln(M, X, Qb, Pp[18] + l * DMODEL, Pp[19] + l * DMODEL, X);
      gemm(X,  Pp[14] + l * DD, Pp[15] + l * DMODEL, Kb, M, DMODEL, DMODEL, 1, 0);
      gemm(Kb, Pp[16] + l * DD, Pp[17] + l * DMODEL, Qb, M, DMODEL, DMODEL, 0, 0);
      ln(M, X, Qb, Pp[20] + l * DMODEL, Pp[21] + l * DMODEL, X);

      if (l < 2) {
        size_t n = (size_t)NB * L * DMODEL;
        inf_shift<<<elts(n), 256, 0, stream>>>(Qb, X, L, -1);
        inf_shift<<<elts(n), 256, 0, stream>>>(Kb, X, L, +1);
        gemm(Qb, Pp[22] + (size_t)(l * 3 + 0) * DD, Pp[23] + l * DMODEL,
             T1, M, DMODEL, DMODEL, 0, 0);
        gemm(X,  Pp[22] + (size_t)(l * 3 + 1) * DD, nullptr,
             T1, M, DMODEL, DMODEL, 0, 1);
        gemm(Kb, Pp[22] + (size_t)(l * 3 + 2) * DD, nullptr,
             T1, M, DMODEL, DMODEL, 0, 1);
        inf_bnstats<<<DMODEL, 256, 0, stream>>>(T1, STAT, M);
        inf_bnelu<<<elts((size_t)M * DMODEL), 256, 0, stream>>>(
            T1, STAT, Pp[24] + l * DMODEL, Pp[25] + l * DMODEL, M);
        inf_maxpool<<<elts((size_t)NB * (L / 2) * DMODEL), 256, 0, stream>>>(X, T1, L);
        L >>= 1;
      }
    }
    ln(NB * L, X, nullptr, Pp[26], Pp[27], ENC);   // L == 256
  }

  // ============================== Decoder ==============================
  {
    const int LD = 512, LE = 256;
    const int MD = NB * LD, ME = NB * LE;
    inf_embed<<<elts((size_t)NB * LD * DMODEL), 256, 0, stream>>>(
        x_dec, mark_dec, Pp[5], XD, LD, 7);

    for (int l = 0; l < 2; l++) {
      // --- self attention (causal) ---
      gemm(XD, Pp[28] + l * DD, Pp[29] + l * DMODEL, Qb, MD, DMODEL, DMODEL, 0, 0);
      gemm(XD, Pp[30] + l * DD, Pp[31] + l * DMODEL, Kb, MD, DMODEL, DMODEL, 0, 0);
      gemm(XD, Pp[32] + l * DD, Pp[33] + l * DMODEL, Vb, MD, DMODEL, DMODEL, 0, 0);
      flash(Qb, Kb, Vb, T1, LD, LD, 1);
      gemm(T1, Pp[34] + l * DD, Pp[35] + l * DMODEL, Qb, MD, DMODEL, DMODEL, 0, 0);
      ln(MD, XD, Qb, Pp[48] + l * DMODEL, Pp[49] + l * DMODEL, XD);
      // --- cross attention ---
      gemm(XD,  Pp[36] + l * DD, Pp[37] + l * DMODEL, Qb, MD, DMODEL, DMODEL, 0, 0);
      gemm(ENC, Pp[38] + l * DD, Pp[39] + l * DMODEL, Kb, ME, DMODEL, DMODEL, 0, 0);
      gemm(ENC, Pp[40] + l * DD, Pp[41] + l * DMODEL, Vb, ME, DMODEL, DMODEL, 0, 0);
      flash(Qb, Kb, Vb, T1, LD, LE, 0);
      gemm(T1, Pp[42] + l * DD, Pp[43] + l * DMODEL, Qb, MD, DMODEL, DMODEL, 0, 0);
      ln(MD, XD, Qb, Pp[50] + l * DMODEL, Pp[51] + l * DMODEL, XD);
      // --- feed forward ---
      gemm(XD, Pp[44] + l * DD, Pp[45] + l * DMODEL, Kb, MD, DMODEL, DMODEL, 1, 0);
      gemm(Kb, Pp[46] + l * DD, Pp[47] + l * DMODEL, Qb, MD, DMODEL, DMODEL, 0, 0);
      ln(MD, XD, Qb, Pp[52] + l * DMODEL, Pp[53] + l * DMODEL, XD);
    }
    ln(MD, XD, nullptr, Pp[54], Pp[55], T1);
    inf_proj<<<(NB * 256 * 7 + 255) / 256, 256, 0, stream>>>(
        T1, Pp[56], Pp[57], (float*)d_out);
  }
}
